// Lensiformer_76166950028187
// MI455X (gfx1250) — compile-verified, matrix operand order checked by reading, not run on hardware
//
#include <hip/hip_runtime.h>
#include <hip/hip_bf16.h>

// ---------------------------------------------------------------------------
// Lensiformer forward for MI455X (gfx1250, wave32, WMMA)
// All heavy GEMMs -> v_wmma_f32_16x16x32_bf16 (bf16 in, f32 accumulate),
// register-blocked 32x64 per wave (8 WMMAs per 32-deep k-step).
// ---------------------------------------------------------------------------

typedef __hip_bfloat16 bf16_t;
typedef __attribute__((ext_vector_type(16))) __bf16 v16bf;
typedef __attribute__((ext_vector_type(8)))  float  v8f;
typedef __attribute__((ext_vector_type(4)))  unsigned int u32x4;

union Frag { u32x4 u[2]; v16bf v; };

// Model constants
#define BATCH 32
#define NTOK  257          // 256 patches + cls
#define DMODEL 512
#define NHEAD 8
#define HDIM  64
#define NLAYER 8
#define MLPD  2048
#define KSPT  320          // 5 * 8 * 8
#define MPATCH (32*256)    // 8192
#define MTOK   (32*257)    // 8224

// ---------------------------------------------------------------------------
// fp32 -> bf16 elementwise convert (weight already in N x K layout)
__global__ __launch_bounds__(256) void k_cvt(const float* __restrict__ src,
                                             bf16_t* __restrict__ dst,
                                             long n) {
  long i = (long)blockIdx.x * 256 + threadIdx.x;
  if (i < n) dst[i] = __float2bfloat16(src[i]);
}

// fp32 (K x N) row-major -> bf16 (N x K) row-major (transpose + convert)
__global__ __launch_bounds__(256) void k_cvt_t(const float* __restrict__ src,
                                               bf16_t* __restrict__ dst,
                                               int K, int N) {
  long i = (long)blockIdx.x * 256 + threadIdx.x;
  if (i >= (long)K * N) return;
  int n = (int)(i % N);
  int k = (int)(i / N);
  dst[(long)n * K + k] = __float2bfloat16(src[i]);
}

// ---------------------------------------------------------------------------
// im2col for shifted-patch tokenization: A (8192 x 320) bf16
// row r = b*256 + py*16 + px ; col c = ch*64 + i*8 + j
// channels: identity + 4 diagonal rolls (torus wrap)
__global__ __launch_bounds__(256) void k_im2col(const float* __restrict__ img,
                                                bf16_t* __restrict__ A) {
  long idx = (long)blockIdx.x * 256 + threadIdx.x;
  if (idx >= (long)MPATCH * KSPT) return;
  int c = (int)(idx % KSPT);
  int r = (int)(idx / KSPT);
  int b  = r >> 8;
  int pi = r & 255;
  int py = pi >> 4, px = pi & 15;
  int ch = c / 64;
  int w64 = c % 64;
  int ii = w64 >> 3, jj = w64 & 7;
  // rolled[h,w] = x[(h - dy) mod 128, (w - dx) mod 128]
  const int dy[5] = {0, 1, -1,  1, -1};
  const int dx[5] = {0, 1,  1, -1, -1};
  int hs = (py * 8 + ii - dy[ch]) & 127;
  int ws = (px * 8 + jj - dx[ch]) & 127;
  A[idx] = __float2bfloat16(img[((long)b * 128 + hs) * 128 + ws]);
}

// ---------------------------------------------------------------------------
// WMMA GEMM: C[M,N] = A[M,K](bf16) * Bt[N,K](bf16)^T (+bias)(+act)(+resid)
// One wave per 32x64 output tile: 2 A-fragments x 4 B-fragments -> 8 WMMAs
// per 32-deep k-step with register-level reuse (1.5 16B-loads per WMMA).
// Requires M%32==0, N%64==0, K%32==0 (true for all GEMMs in this model).
// act: 0=none, 1=exact GELU.
__global__ __launch_bounds__(256) void k_wmma_gemm(
    const bf16_t* __restrict__ A, const bf16_t* __restrict__ Bt,
    const float* __restrict__ bias, const float* __restrict__ resid,
    float* __restrict__ outF, bf16_t* __restrict__ outB,
    int M, int N, int K, int act) {
  int wave = (int)((blockIdx.x * 256 + threadIdx.x) >> 5);
  int lane = threadIdx.x & 31;
  int tilesN = N >> 6;                    // tiles of 64 columns
  int tm = wave / tilesN;
  int tn = wave - tm * tilesN;
  if (tm * 32 >= M) return;               // wave-uniform exit

  int hl  = lane >> 4;                    // half of wave: K offset +8
  int l16 = lane & 15;
  const bf16_t* ap0 = A  + (long)(tm * 32 + l16) * K + hl * 8;
  const bf16_t* ap1 = ap0 + (long)16 * K;
  const bf16_t* bp  = Bt + (long)(tn * 64 + l16) * K + hl * 8;
  const long bstep = (long)16 * K;        // next 16-column B fragment

  v8f acc[2][4];
  #pragma unroll
  for (int i = 0; i < 2; ++i)
    #pragma unroll
    for (int j = 0; j < 4; ++j) acc[i][j] = (v8f){};

  for (int k0 = 0; k0 < K; k0 += 32) {
    Frag a[2], b[4];
    a[0].u[0] = *(const u32x4*)(ap0 + k0);
    a[0].u[1] = *(const u32x4*)(ap0 + k0 + 16);
    a[1].u[0] = *(const u32x4*)(ap1 + k0);
    a[1].u[1] = *(const u32x4*)(ap1 + k0 + 16);
    #pragma unroll
    for (int j = 0; j < 4; ++j) {
      b[j].u[0] = *(const u32x4*)(bp + j * bstep + k0);
      b[j].u[1] = *(const u32x4*)(bp + j * bstep + k0 + 16);
    }
    __builtin_prefetch(ap0 + k0 + 128, 0, 3);   // near-cache speculative
    __builtin_prefetch(ap1 + k0 + 128, 0, 3);
    __builtin_prefetch(bp + k0 + 128, 0, 3);
    __builtin_prefetch(bp + 3 * bstep + k0 + 128, 0, 3);
    #pragma unroll
    for (int i = 0; i < 2; ++i)
      #pragma unroll
      for (int j = 0; j < 4; ++j)
        acc[i][j] = __builtin_amdgcn_wmma_f32_16x16x32_bf16(
            false, a[i].v, false, b[j].v, (short)0, acc[i][j], false, false);
  }

  #pragma unroll
  for (int j = 0; j < 4; ++j) {
    int col = tn * 64 + j * 16 + l16;
    float bv = bias ? bias[col] : 0.f;
    #pragma unroll
    for (int i = 0; i < 2; ++i) {
      #pragma unroll
      for (int r = 0; r < 8; ++r) {
        int row = tm * 32 + i * 16 + r + hl * 8;  // lanes 16-31 -> M+8
        float x = acc[i][j][r] + bv;
        if (act == 1) x = 0.5f * x * (1.f + erff(x * 0.70710678118f));
        long idx = (long)row * N + col;
        if (resid) x += resid[idx];
        if (outF) outF[idx] = x;
        if (outB) outB[idx] = __float2bfloat16(x);
      }
    }
  }
}

// ---------------------------------------------------------------------------
// LayerNorm over D=512, one wave per row; optional f32 and bf16 outputs
__global__ __launch_bounds__(256) void k_ln(const float* __restrict__ x,
                                            const float* __restrict__ g,
                                            const float* __restrict__ be,
                                            float* __restrict__ outF,
                                            bf16_t* __restrict__ outB,
                                            int rows, int obStride, int obOff) {
  int wid  = (int)((blockIdx.x * 256 + threadIdx.x) >> 5);
  int lane = threadIdx.x & 31;
  if (wid >= rows) return;
  const float* xr = x + (long)wid * DMODEL;
  float v[16];
  float s = 0.f;
  #pragma unroll
  for (int i = 0; i < 16; ++i) { v[i] = xr[lane + i * 32]; s += v[i]; }
  for (int o = 16; o; o >>= 1) s += __shfl_xor(s, o, 32);
  float mu = s * (1.f / DMODEL);
  float q = 0.f;
  #pragma unroll
  for (int i = 0; i < 16; ++i) { float d = v[i] - mu; q += d * d; }
  for (int o = 16; o; o >>= 1) q += __shfl_xor(q, o, 32);
  float rstd = rsqrtf(q * (1.f / DMODEL) + 1e-5f);
  #pragma unroll
  for (int i = 0; i < 16; ++i) {
    int d = lane + i * 32;
    float y = (v[i] - mu) * rstd * g[d] + be[d];
    if (outF) outF[(long)wid * DMODEL + d] = y;
    if (outB) outB[(long)wid * obStride + obOff + d] = __float2bfloat16(y);
  }
}

// ---------------------------------------------------------------------------
// Token assembly: sigmoid-gate fusion of the two SPT streams, prepend cls,
// add positional embedding. tok is (B, 257, 512) f32.
__global__ __launch_bounds__(256) void k_tokens(const float* __restrict__ srcF,
                                                const float* __restrict__ origF,
                                                const float* __restrict__ gateL,
                                                const float* __restrict__ cls,
                                                const float* __restrict__ pos,
                                                float* __restrict__ tok) {
  long idx = (long)blockIdx.x * 256 + threadIdx.x;
  if (idx >= (long)BATCH * NTOK * DMODEL) return;
  int d = (int)(idx % DMODEL);
  int n = (int)((idx / DMODEL) % NTOK);
  int b = (int)(idx / ((long)DMODEL * NTOK));
  float p = pos[(long)n * DMODEL + d];
  float val;
  if (n == 0) {
    val = cls[d] + p;
  } else {
    long r = ((long)b * 256 + (n - 1)) * DMODEL + d;
    float gt = 1.f / (1.f + __expf(-gateL[r]));
    val = gt * srcF[r] + (1.f - gt) * origF[r] + p;
  }
  tok[idx] = val;
}

// ---------------------------------------------------------------------------
// Attention: one wave per (b, h, query n). N=257 keys, HD=64.
// Diagonal masked to -inf before softmax. Output written bf16 at col h*64.
__global__ __launch_bounds__(256) void k_attn(const float* __restrict__ qkv,
                                              const float* __restrict__ temp,
                                              bf16_t* __restrict__ o) {
  __shared__ float shq[8][64];
  __shared__ float shp[8][264];
  int wib  = threadIdx.x >> 5;
  int lane = threadIdx.x & 31;
  int wid  = blockIdx.x * 8 + wib;          // 65792 waves, 8224 full blocks
  int n = wid % NTOK;
  int h = (wid / NTOK) & (NHEAD - 1);
  int b = wid / (NTOK * NHEAD);

  long rq = ((long)(b * NTOK + n)) * (3 * DMODEL) + h * HDIM;
  shq[wib][lane]      = qkv[rq + lane];
  shq[wib][lane + 32] = qkv[rq + lane + 32];
  __syncthreads();

  float tv = temp[h];
  float sc[9];
  float lmax = -3.4e38f;
  int cnt = 0;
  for (int m = lane; m < NTOK; m += 32) {
    const float* kr = qkv + ((long)(b * NTOK + m)) * (3 * DMODEL) + DMODEL + h * HDIM;
    float s = 0.f;
    #pragma unroll 8
    for (int d = 0; d < HDIM; ++d) s += shq[wib][d] * kr[d];
    s *= tv;
    if (m == n) s = -3.0e38f;               // masked diagonal
    sc[cnt++] = s;
    lmax = fmaxf(lmax, s);
  }
  for (int off = 16; off; off >>= 1) lmax = fmaxf(lmax, __shfl_xor(lmax, off, 32));
  float lsum = 0.f;
  cnt = 0;
  for (int m = lane; m < NTOK; m += 32) {
    float e = __expf(sc[cnt] - lmax);
    sc[cnt++] = e;
    lsum += e;
  }
  for (int off = 16; off; off >>= 1) lsum += __shfl_xor(lsum, off, 32);
  float inv = 1.f / lsum;
  cnt = 0;
  for (int m = lane; m < NTOK; m += 32) shp[wib][m] = sc[cnt++] * inv;
  __syncthreads();

  float a0 = 0.f, a1 = 0.f;
  for (int m = 0; m < NTOK; ++m) {
    const float* vr = qkv + ((long)(b * NTOK + m)) * (3 * DMODEL) + 2 * DMODEL + h * HDIM;
    float p = shp[wib][m];
    a0 += p * vr[lane];
    a1 += p * vr[lane + 32];
  }
  long ro = ((long)(b * NTOK + n)) * DMODEL + h * HDIM;
  o[ro + lane]      = __float2bfloat16(a0);
  o[ro + lane + 32] = __float2bfloat16(a1);
}

// ---------------------------------------------------------------------------
// Final: LN(cls row) then 512x3 head. One wave per batch element.
__global__ __launch_bounds__(256) void k_head(const float* __restrict__ tok,
                                              const float* __restrict__ ng,
                                              const float* __restrict__ nb,
                                              const float* __restrict__ hw,
                                              const float* __restrict__ hb,
                                              float* __restrict__ out) {
  int wid  = (int)((blockIdx.x * 256 + threadIdx.x) >> 5);
  int lane = threadIdx.x & 31;
  if (wid >= BATCH) return;
  const float* xr = tok + (long)wid * NTOK * DMODEL;   // cls row
  float v[16];
  float s = 0.f;
  #pragma unroll
  for (int i = 0; i < 16; ++i) { v[i] = xr[lane + i * 32]; s += v[i]; }
  for (int o = 16; o; o >>= 1) s += __shfl_xor(s, o, 32);
  float mu = s * (1.f / DMODEL);
  float q = 0.f;
  #pragma unroll
  for (int i = 0; i < 16; ++i) { float d = v[i] - mu; q += d * d; }
  for (int o = 16; o; o >>= 1) q += __shfl_xor(q, o, 32);
  float rstd = rsqrtf(q * (1.f / DMODEL) + 1e-5f);
  float acc0 = 0.f, acc1 = 0.f, acc2 = 0.f;
  #pragma unroll
  for (int i = 0; i < 16; ++i) {
    int d = lane + i * 32;
    float y = (v[i] - mu) * rstd * ng[d] + nb[d];
    acc0 += y * hw[d * 3 + 0];
    acc1 += y * hw[d * 3 + 1];
    acc2 += y * hw[d * 3 + 2];
  }
  for (int o = 16; o; o >>= 1) {
    acc0 += __shfl_xor(acc0, o, 32);
    acc1 += __shfl_xor(acc1, o, 32);
    acc2 += __shfl_xor(acc2, o, 32);
  }
  if (lane == 0) {
    out[wid * 3 + 0] = acc0 + hb[0];
    out[wid * 3 + 1] = acc1 + hb[1];
    out[wid * 3 + 2] = acc2 + hb[2];
  }
}

// ---------------------------------------------------------------------------
extern "C" void kernel_launch(void* const* d_in, const int* in_sizes, int n_in,
                              void* d_out, int out_size, void* d_ws, size_t ws_size,
                              hipStream_t stream) {
  (void)in_sizes; (void)n_in; (void)out_size; (void)ws_size;
  const float* image     = (const float*)d_in[0];
  const float* ssw       = (const float*)d_in[1];
  const float* ssb       = (const float*)d_in[2];
  const float* ssg       = (const float*)d_in[3];
  const float* ssbeta    = (const float*)d_in[4];
  const float* sow       = (const float*)d_in[5];
  const float* sob       = (const float*)d_in[6];
  const float* sog       = (const float*)d_in[7];
  const float* sobeta    = (const float*)d_in[8];
  const float* fw        = (const float*)d_in[9];
  const float* fb        = (const float*)d_in[10];
  const float* cls_token = (const float*)d_in[11];
  const float* pos_embed = (const float*)d_in[12];
  const float* ln1g      = (const float*)d_in[13];
  const float* ln1b      = (const float*)d_in[14];
  const float* qkvw      = (const float*)d_in[15];
  const float* qkvb      = (const float*)d_in[16];
  const float* projw     = (const float*)d_in[17];
  const float* projb     = (const float*)d_in[18];
  const float* temp      = (const float*)d_in[19];
  const float* ln2g      = (const float*)d_in[20];
  const float* ln2b      = (const float*)d_in[21];
  const float* w1        = (const float*)d_in[22];
  const float* b1        = (const float*)d_in[23];
  const float* w2        = (const float*)d_in[24];
  const float* b2        = (const float*)d_in[25];
  const float* ng        = (const float*)d_in[26];
  const float* nb        = (const float*)d_in[27];
  const float* hw        = (const float*)d_in[28];
  const float* hb        = (const float*)d_in[29];

  // bump allocator over workspace (256B aligned)
  char* ws = (char*)d_ws;
  size_t off = 0;
  auto alloc = [&](size_t bytes) -> void* {
    off = (off + 255) & ~(size_t)255;
    void* p = ws + off;
    off += bytes;
    return p;
  };

  // bf16 weights (transposed to N x K where needed)
  bf16_t* sswB  = (bf16_t*)alloc((size_t)DMODEL * KSPT * 2);        // already (N,K)
  bf16_t* sowB  = (bf16_t*)alloc((size_t)DMODEL * KSPT * 2);
  bf16_t* fwT   = (bf16_t*)alloc((size_t)DMODEL * (2 * DMODEL) * 2);
  bf16_t* qkvwT = (bf16_t*)alloc((size_t)NLAYER * 3 * DMODEL * DMODEL * 2);
  bf16_t* projwT= (bf16_t*)alloc((size_t)NLAYER * DMODEL * DMODEL * 2);
  bf16_t* w1T   = (bf16_t*)alloc((size_t)NLAYER * MLPD * DMODEL * 2);
  bf16_t* w2T   = (bf16_t*)alloc((size_t)NLAYER * DMODEL * MLPD * 2);
  // activations
  bf16_t* Aim   = (bf16_t*)alloc((size_t)MPATCH * KSPT * 2);
  float*  srcRaw= (float*)alloc((size_t)MPATCH * DMODEL * 4);
  float*  origRaw=(float*)alloc((size_t)MPATCH * DMODEL * 4);
  float*  srcF  = (float*)alloc((size_t)MPATCH * DMODEL * 4);
  float*  origF = (float*)alloc((size_t)MPATCH * DMODEL * 4);
  bf16_t* catB  = (bf16_t*)alloc((size_t)MPATCH * (2 * DMODEL) * 2);
  float*  gateL = (float*)alloc((size_t)MPATCH * DMODEL * 4);
  float*  tok   = (float*)alloc((size_t)MTOK * DMODEL * 4);
  bf16_t* xlnB  = (bf16_t*)alloc((size_t)MTOK * DMODEL * 2);
  float*  qkvF  = (float*)alloc((size_t)MTOK * 3 * DMODEL * 4);
  bf16_t* oB    = (bf16_t*)alloc((size_t)MTOK * DMODEL * 2);
  bf16_t* h1B   = (bf16_t*)alloc((size_t)MTOK * MLPD * 2);

  auto blocks_for = [](long n) { return (unsigned)((n + 255) / 256); };
  auto gemm = [&](const bf16_t* A, const bf16_t* Bt, const float* bias,
                  const float* resid, float* outF, bf16_t* outB,
                  int M, int N, int K, int act) {
    long tiles = (long)(M / 32) * (N / 64);          // one wave per 32x64 tile
    unsigned blocks = (unsigned)((tiles + 7) / 8);   // 8 waves per block
    k_wmma_gemm<<<blocks, 256, 0, stream>>>(A, Bt, bias, resid, outF, outB,
                                            M, N, K, act);
  };

  // ---- weight conversion (recomputed each call: deterministic) ----
  k_cvt<<<blocks_for((long)DMODEL * KSPT), 256, 0, stream>>>(ssw, sswB, (long)DMODEL * KSPT);
  k_cvt<<<blocks_for((long)DMODEL * KSPT), 256, 0, stream>>>(sow, sowB, (long)DMODEL * KSPT);
  k_cvt_t<<<blocks_for((long)2 * DMODEL * DMODEL), 256, 0, stream>>>(fw, fwT, 2 * DMODEL, DMODEL);
  for (int l = 0; l < NLAYER; ++l) {
    k_cvt_t<<<blocks_for((long)DMODEL * 3 * DMODEL), 256, 0, stream>>>(
        qkvw + (long)l * DMODEL * 3 * DMODEL, qkvwT + (long)l * 3 * DMODEL * DMODEL,
        DMODEL, 3 * DMODEL);
    k_cvt_t<<<blocks_for((long)DMODEL * DMODEL), 256, 0, stream>>>(
        projw + (long)l * DMODEL * DMODEL, projwT + (long)l * DMODEL * DMODEL,
        DMODEL, DMODEL);
    k_cvt_t<<<blocks_for((long)DMODEL * MLPD), 256, 0, stream>>>(
        w1 + (long)l * DMODEL * MLPD, w1T + (long)l * MLPD * DMODEL, DMODEL, MLPD);
    k_cvt_t<<<blocks_for((long)MLPD * DMODEL), 256, 0, stream>>>(
        w2 + (long)l * MLPD * DMODEL, w2T + (long)l * DMODEL * MLPD, MLPD, DMODEL);
  }

  // ---- shifted-patch tokenization (shared im2col, two WMMA GEMMs) ----
  k_im2col<<<blocks_for((long)MPATCH * KSPT), 256, 0, stream>>>(image, Aim);
  gemm(Aim, sswB, ssb, nullptr, srcRaw, nullptr, MPATCH, DMODEL, KSPT, 0);
  gemm(Aim, sowB, sob, nullptr, origRaw, nullptr, MPATCH, DMODEL, KSPT, 0);
  k_ln<<<blocks_for((long)MPATCH * 32), 256, 0, stream>>>(
      srcRaw, ssg, ssbeta, srcF, catB, MPATCH, 2 * DMODEL, 0);
  k_ln<<<blocks_for((long)MPATCH * 32), 256, 0, stream>>>(
      origRaw, sog, sobeta, origF, catB, MPATCH, 2 * DMODEL, DMODEL);
  // gate logits = [src, orig] @ fw + fb
  gemm(catB, fwT, fb, nullptr, gateL, nullptr, MPATCH, DMODEL, 2 * DMODEL, 0);
  k_tokens<<<blocks_for((long)BATCH * NTOK * DMODEL), 256, 0, stream>>>(
      srcF, origF, gateL, cls_token, pos_embed, tok);

  // ---- transformer blocks ----
  for (int l = 0; l < NLAYER; ++l) {
    k_ln<<<blocks_for((long)MTOK * 32), 256, 0, stream>>>(
        tok, ln1g + l * DMODEL, ln1b + l * DMODEL, nullptr, xlnB, MTOK, DMODEL, 0);
    gemm(xlnB, qkvwT + (long)l * 3 * DMODEL * DMODEL, qkvb + l * 3 * DMODEL,
         nullptr, qkvF, nullptr, MTOK, 3 * DMODEL, DMODEL, 0);
    k_attn<<<MTOK, 256, 0, stream>>>(qkvF, temp + l * NHEAD, oB);  // 8224 blocks, 8 waves each
    gemm(oB, projwT + (long)l * DMODEL * DMODEL, projb + l * DMODEL,
         tok, tok, nullptr, MTOK, DMODEL, DMODEL, 0);
    k_ln<<<blocks_for((long)MTOK * 32), 256, 0, stream>>>(
        tok, ln2g + l * DMODEL, ln2b + l * DMODEL, nullptr, xlnB, MTOK, DMODEL, 0);
    gemm(xlnB, w1T + (long)l * MLPD * DMODEL, b1 + l * MLPD,
         nullptr, nullptr, h1B, MTOK, MLPD, DMODEL, 1 /*exact GELU*/);
    gemm(h1B, w2T + (long)l * DMODEL * MLPD, b2 + l * DMODEL,
         tok, tok, nullptr, MTOK, DMODEL, MLPD, 0);
  }

  // ---- head ----
  k_head<<<4, 256, 0, stream>>>(tok, ng, nb, hw, hb, (float*)d_out);
}
